// Encoder_overall_71098888618254
// MI455X (gfx1250) — matrix operand order repaired and checked by reading, not compiled, not run on hardware
//
#include <hip/hip_runtime.h>

// CDNA5 / gfx1250: wave32, WMMA bf16 16x16x32, fp32 accumulate.
// - Fragment-linear LDS: each lane fetches operands as contiguous 16B chunks
//   (ds_load_b128, conflict-free; lane stride 16B inside each chunk plane).
// - Full K-tiles use check-free fetches; the single partial tail tile is
//   peeled (clamped loads; B supplies exact zeros for k>=K so clamped A
//   garbage multiplies against zero).
// - Register-prefetch pipeline: tile kt+1 global loads overlap tile kt WMMAs.
// - Native bf16 converts (v_cvt_pk_bf16_f32).

typedef __attribute__((ext_vector_type(16))) __bf16 v16bf;
typedef __attribute__((ext_vector_type(8)))  __bf16 bf16x8;
typedef __attribute__((ext_vector_type(8)))  float  v8f;
typedef __attribute__((ext_vector_type(8)))  unsigned short u16x8;

#define BM 128   // rows per block (8 waves x 16)
#define BN 64    // cols per block (4 wmma tiles of 16 per wave)
#define BK 32    // k-step (wmma bf16 K)  -- requires K >= BK (all GEMMs here: K>=64)
#define NT 256   // threads per block (8 wave32)

__device__ __forceinline__ int imin(int a, int b) { return a < b ? a : b; }

union Frag { u16x8 h[2]; bf16x8 v[2]; v16bf b; };

// One 16x16 N-tile: 2x ds_load_b128 for B fragment + WMMA.
// T is a template parameter so the reuse_a hint is a frontend constant.
template <int T>
__device__ __forceinline__ v8f wmma_tile(v16bf a, const unsigned short* __restrict__ Bs,
                                         int lane, v8f c) {
  Frag fb;
  fb.h[0] = *(const u16x8*)&Bs[((T * 2 + 0) * 32 + lane) * 8];
  fb.h[1] = *(const u16x8*)&Bs[((T * 2 + 1) * 32 + lane) * 8];
  return __builtin_amdgcn_wmma_f32_16x16x32_bf16(
      /*neg_a=*/false, a, /*neg_b=*/false, fb.b,
      /*c_mod=*/(short)0, c, /*reuse_a=*/(T < 3), /*reuse_b=*/false);
}

// AMODE 0: A = A0 (row-major, lda)
// AMODE 1: A = w0*A0 + w1*A1 + b   (conv1x1 adjacency fusion)
// AMODE 2: A = concat3 along K: A0[(k>>6)*catStride + m*64 + (k&63)]
template <int AMODE>
__device__ __forceinline__ void fetchA_fast(float va[16],
    const float* __restrict__ A0, const float* __restrict__ A1,
    float w0, float w1, float badd,
    int mclamp, int kA, int lda, long catStride) {
  const float* src = (AMODE == 2)
      ? (A0 + (long)(kA >> 6) * catStride + (long)mclamp * 64 + (kA & 63))
      : (A0 + (long)mclamp * lda + kA);
#pragma unroll
  for (int q = 0; q < 4; ++q) {
    const float4 v = ((const float4*)src)[q];
    va[4 * q + 0] = v.x; va[4 * q + 1] = v.y; va[4 * q + 2] = v.z; va[4 * q + 3] = v.w;
  }
  if (AMODE == 1) {
    const float* src1 = A1 + (long)mclamp * lda + kA;
#pragma unroll
    for (int q = 0; q < 4; ++q) {
      const float4 v = ((const float4*)src1)[q];
      va[4 * q + 0] = w0 * va[4 * q + 0] + w1 * v.x + badd;
      va[4 * q + 1] = w0 * va[4 * q + 1] + w1 * v.y + badd;
      va[4 * q + 2] = w0 * va[4 * q + 2] + w1 * v.z + badd;
      va[4 * q + 3] = w0 * va[4 * q + 3] + w1 * v.w + badd;
    }
  }
}

template <int AMODE>
__device__ __forceinline__ void fetchA_tail(float va[16],
    const float* __restrict__ A0, const float* __restrict__ A1,
    float w0, float w1, float badd,
    int mclamp, int kA, int K, int lda, long catStride) {
  // clamp k; values at k>=K are don't-care (B holds zeros there)
#pragma unroll
  for (int j = 0; j < 16; ++j) {
    const int kc = imin(kA + j, K - 1);
    if (AMODE == 0)      va[j] = A0[(long)mclamp * lda + kc];
    else if (AMODE == 1) va[j] = w0 * A0[(long)mclamp * lda + kc] +
                                 w1 * A1[(long)mclamp * lda + kc] + badd;
    else                 va[j] = A0[(long)(kc >> 6) * catStride + (long)mclamp * 64 + (kc & 63)];
  }
}

template <int LDB>
__device__ __forceinline__ void fetchB_fast(float vb[8], const float* __restrict__ Bcol,
                                            int k0b) {
  const float* p = Bcol + (long)k0b * LDB;
#pragma unroll
  for (int j = 0; j < 8; ++j) vb[j] = p[j * LDB];
}

template <int LDB>
__device__ __forceinline__ void fetchB_tail(float vb[8], const float* __restrict__ Bcol,
                                            int k0b, int K) {
#pragma unroll
  for (int j = 0; j < 8; ++j) {
    const int k = k0b + j;
    const int kc = imin(k, K - 1);
    const float v = Bcol[(long)kc * LDB];
    vb[j] = (k < K) ? v : 0.f;  // exact zeros for padded k (protects accumulation)
  }
}

// C[M,N] = Afused[M,K] @ B[K,N] (+ bias[N])
template <int AMODE, int LDB>
__global__ __launch_bounds__(NT)
void gemm_wmma_bf16(const float* __restrict__ A0, const float* __restrict__ A1,
                    const float* __restrict__ wvec, const float* __restrict__ bvec,
                    const float* __restrict__ B, const float* __restrict__ bias,
                    float* __restrict__ C,
                    int M, int N, int K, int lda, int ldc, long catStride) {
  // Fragment-linear LDS: [group][chunk][lane][8 bf16]
  __shared__ unsigned short As[(BM / 16) * 2 * 32 * 8];  // 8KB
  __shared__ unsigned short Bs[(BN / 16) * 2 * 32 * 8];  // 4KB

  const int tid  = threadIdx.x;
  const int lane = tid & 31;
  const int wave = tid >> 5;
  const int m0 = blockIdx.x * BM;
  const int n0 = blockIdx.y * BN;

  float w0 = 0.f, w1 = 0.f, badd = 0.f;
  if (AMODE == 1) { w0 = wvec[0]; w1 = wvec[1]; badd = bvec[0]; }

  // ---- staging roles ------------------------------------------------------
  // A: thread owns (row, 16 consecutive k): row = tid>>1, half = tid&1.
  //    Frag map: lane=(r&15)+16*((k>>3)&1), chunk c=(k>=16), slot=k&7.
  const int arow  = tid >> 1;
  const int ahalf = tid & 1;
  unsigned short* aDstLo = &As[(((arow >> 4) * 2 + ahalf) * 32 + (arow & 15)) * 8];
  unsigned short* aDstHi = aDstLo + 16 * 8;
  // B: thread owns (col, 8 consecutive k): n = tid&63, kb = (tid>>6)*8.
  //    Frag map: t=n>>4, lane=(n&15)+16*((k>>3)&1), c=(k>=16), slot=k&7.
  const int bn  = tid & 63;
  const int bkb = (tid >> 6) << 3;
  unsigned short* bDst =
      &Bs[(((bn >> 4) * 2 + ((bkb >= 16) ? 1 : 0)) * 32 + (bn & 15) + (((bkb >> 3) & 1) ? 16 : 0)) * 8];

  const int mclamp = imin(m0 + arow, M - 1);              // garbage rows never stored
  const float* Bcol = B + imin(n0 + bn, N - 1);           // garbage cols never stored

  v8f acc[4];
#pragma unroll
  for (int t = 0; t < 4; ++t)
#pragma unroll
    for (int v = 0; v < 8; ++v) acc[t][v] = 0.f;

  const int fullTiles = K / BK;           // >= 1 for all launches here
  const bool hasTail  = (K - fullTiles * BK) != 0;

  float va[16], vb[8];

  auto stage = [&]() {  // convert (v_cvt_pk_bf16_f32) + 3x ds_store_b128
    bf16x8 alo, ahi, bvv;
#pragma unroll
    for (int j = 0; j < 8; ++j) {
      alo[j] = (__bf16)va[j];
      ahi[j] = (__bf16)va[j + 8];
      bvv[j] = (__bf16)vb[j];
    }
    *(bf16x8*)aDstLo = alo;
    *(bf16x8*)aDstHi = ahi;
    *(bf16x8*)bDst   = bvv;
  };
  auto compute = [&]() {  // 2x ds_load_b128 for A, 4 WMMAs sharing A (reuse_a)
    Frag fa;
    fa.h[0] = *(const u16x8*)&As[((wave * 2 + 0) * 32 + lane) * 8];
    fa.h[1] = *(const u16x8*)&As[((wave * 2 + 1) * 32 + lane) * 8];
    acc[0] = wmma_tile<0>(fa.b, Bs, lane, acc[0]);
    acc[1] = wmma_tile<1>(fa.b, Bs, lane, acc[1]);
    acc[2] = wmma_tile<2>(fa.b, Bs, lane, acc[2]);
    acc[3] = wmma_tile<3>(fa.b, Bs, lane, acc[3]);
  };

  // ---- software pipeline: prefetch tile 0 (always full) -------------------
  fetchA_fast<AMODE>(va, A0, A1, w0, w1, badd, mclamp, ahalf * 16, lda, catStride);
  fetchB_fast<LDB>(vb, Bcol, bkb);

  for (int kt = 0; kt < fullTiles; ++kt) {
    stage();
    __syncthreads();
    // issue next tile's global loads; they overlap the WMMAs below
    if (kt + 1 < fullTiles) {       // uniform scalar branch, check-free fetch
      const int k0n = (kt + 1) * BK;
      fetchA_fast<AMODE>(va, A0, A1, w0, w1, badd, mclamp, k0n + ahalf * 16, lda, catStride);
      fetchB_fast<LDB>(vb, Bcol, k0n + bkb);
    } else if (hasTail) {           // runs once
      const int k0n = fullTiles * BK;
      fetchA_tail<AMODE>(va, A0, A1, w0, w1, badd, mclamp, k0n + ahalf * 16, K, lda, catStride);
      fetchB_tail<LDB>(vb, Bcol, k0n + bkb, K);
    }
    compute();
    __syncthreads();
  }
  if (hasTail) {                    // peeled partial tile
    stage();
    __syncthreads();
    compute();
  }

  // ---- epilogue: D layout (lane<16: M=v, lane>=16: M=v+8; N=lane&15) -------
  const int mrow = m0 + wave * 16 + ((lane < 16) ? 0 : 8);
#pragma unroll
  for (int t = 0; t < 4; ++t) {
    const int n = n0 + (lane & 15) + 16 * t;
    if (n >= N) continue;
    const float bb = (bias != nullptr) ? bias[n] : 0.f;
#pragma unroll
    for (int v = 0; v < 8; ++v) {
      const int m = mrow + v;
      if (m < M) C[(long)m * ldc + n] = acc[t][v] + bb;
    }
  }
}

extern "C" void kernel_launch(void* const* d_in, const int* in_sizes, int n_in,
                              void* d_out, int out_size, void* d_ws, size_t ws_size,
                              hipStream_t stream) {
  (void)in_sizes; (void)n_in; (void)out_size; (void)ws_size;
  const int N = 6000, DIN = 3000, DOUT = 64;

  const float* feat[3]  = {(const float*)d_in[0],  (const float*)d_in[1],  (const float*)d_in[2]};
  const float* adjsp[3] = {(const float*)d_in[3],  (const float*)d_in[5],  (const float*)d_in[7]};
  const float* adjft[3] = {(const float*)d_in[4],  (const float*)d_in[6],  (const float*)d_in[8]};
  const float* convw[3] = {(const float*)d_in[9],  (const float*)d_in[11], (const float*)d_in[13]};
  const float* convb[3] = {(const float*)d_in[10], (const float*)d_in[12], (const float*)d_in[14]};
  const float* Wenc[3]  = {(const float*)d_in[15], (const float*)d_in[16], (const float*)d_in[17]};
  const float* Wdec[3]  = {(const float*)d_in[18], (const float*)d_in[19], (const float*)d_in[20]};
  const float* mw1 = (const float*)d_in[21];
  const float* mb1 = (const float*)d_in[22];
  const float* mw2 = (const float*)d_in[23];
  const float* mb2 = (const float*)d_in[24];

  float* out = (float*)d_out;
  const long latSz = (long)N * DOUT;           // 384000
  float* lat      = out;                       // lat1|lat2|lat3
  float* combined = out + 3 * latSz;
  float* rec      = out + 4 * latSz;           // rec1|rec2|rec3, each N*DIN

  float* ws   = (float*)d_ws;
  float* tmpE = ws;                // 3 * latSz : feat_i @ Wenc_i
  float* h    = ws + 3 * latSz;    // 1 * latSz : MLP hidden
  float* tD   = ws + 4 * latSz;    // 3 * latSz : adj_sp_i @ combined

  const dim3 blk(NT);
  const dim3 gN64((N + BM - 1) / BM, 1);                    // 47 x 1
  const dim3 gRec((N + BM - 1) / BM, (DIN + BN - 1) / BN);  // 47 x 47

  // 1) tmpE_i = feat_i @ W_enc_i           [6000,3000]@[3000,64]
  for (int i = 0; i < 3; ++i)
    gemm_wmma_bf16<0, 64><<<gN64, blk, 0, stream>>>(
        feat[i], nullptr, nullptr, nullptr, Wenc[i], nullptr,
        tmpE + i * latSz, N, DOUT, DIN, DIN, DOUT, 0);

  // 2) lat_i = (w0*adj_sp + w1*adj_ft + b) @ tmpE_i   (conv1x1 fused into A-load)
  for (int i = 0; i < 3; ++i)
    gemm_wmma_bf16<1, 64><<<gN64, blk, 0, stream>>>(
        adjsp[i], adjft[i], convw[i], convb[i], tmpE + i * latSz, nullptr,
        lat + i * latSz, N, DOUT, N, N, DOUT, 0);

  // 3) h = concat(lat1,lat2,lat3) @ mlp_w1 + b1       [6000,192]@[192,64]
  gemm_wmma_bf16<2, 64><<<gN64, blk, 0, stream>>>(
      lat, nullptr, nullptr, nullptr, mw1, mb1,
      h, N, DOUT, 3 * DOUT, DOUT, DOUT, latSz);

  // 4) combined = h @ mlp_w2 + b2                     [6000,64]@[64,64]
  gemm_wmma_bf16<0, 64><<<gN64, blk, 0, stream>>>(
      h, nullptr, nullptr, nullptr, mw2, mb2,
      combined, N, DOUT, DOUT, DOUT, DOUT, 0);

  // 5) tD_i = adj_sp_i @ combined                     [6000,6000]@[6000,64]
  for (int i = 0; i < 3; ++i)
    gemm_wmma_bf16<0, 64><<<gN64, blk, 0, stream>>>(
        adjsp[i], nullptr, nullptr, nullptr, combined, nullptr,
        tD + i * latSz, N, DOUT, N, N, DOUT, 0);

  // 6) rec_i = tD_i @ W_dec_i  (reassociated decoder) [6000,64]@[64,3000]
  for (int i = 0; i < 3; ++i)
    gemm_wmma_bf16<0, 3000><<<gRec, blk, 0, stream>>>(
        tD + i * latSz, nullptr, nullptr, nullptr, Wdec[i], nullptr,
        rec + (long)i * N * DIN, N, DIN, DOUT, DOUT, DIN, 0);
}
